// MMHA_14491219657401
// MI455X (gfx1250) — matrix operand order repaired and checked by reading, not compiled, not run on hardware
//
#include <hip/hip_runtime.h>
#include <hip/hip_bf16.h>

// ---------------------------------------------------------------------------
// MHA with RoPE, causal flash attention.  bf16 WMMA (v_wmma_f32_16x16x32_bf16,
// f32 accumulate) everywhere; 256-thread blocks = 8 wave32s; each wave computes
// a 16x128 output strip (8 WMMA accumulators).
// ---------------------------------------------------------------------------

#define BATCH  2
#define CTX    2048
#define D_IN   2048
#define D_OUT  2048
#define NH     16
#define HD     128
#define BS     (BATCH * CTX)   // 4096 rows

typedef unsigned short u16;
typedef __attribute__((ext_vector_type(16))) __bf16 v16bf;
typedef __attribute__((ext_vector_type(8)))  float  v8f;

// -------------------------- helpers ---------------------------------------

__device__ __forceinline__ u16 f32_to_bf16(float f) {
    unsigned u = __builtin_bit_cast(unsigned, f);
    u += 0x7FFFu + ((u >> 16) & 1u);   // round-to-nearest-even
    return (u16)(u >> 16);
}

// A-matrix 16x32 bf16 fragment (lanes 0-15: K={0..7,16..23}; lanes 16-31: +8)
__device__ __forceinline__ v16bf load_a_frag(const u16* tile, int ld, int lane) {
    const int row  = lane & 15;
    const int koff = (lane >> 4) * 8;
    const u16* p = tile + (size_t)row * ld + koff;
    union { uint4 u[2]; v16bf v; } t;
    t.u[0] = *(const uint4*)(p);
    t.u[1] = *(const uint4*)(p + 16);
    return t.v;
}

// B-matrix 32x16 bf16 fragment from an N-major [n][k] tile
// (lanes 0-15: K={0..7,16..23} of column n; lanes 16-31: +8)
__device__ __forceinline__ v16bf load_b_frag(const u16* tile, int ld, int kbase, int lane) {
    const int col   = lane & 15;
    const int khalf = (lane >> 4) * 8;
    const u16* p = tile + (size_t)col * ld + kbase + khalf;
    union { uint4 u[2]; v16bf v; } t;
    t.u[0] = *(const uint4*)(p);
    t.u[1] = *(const uint4*)(p + 16);
    return t.v;
}

__device__ __forceinline__ v8f wmma_bf16(v16bf a, v16bf b, v8f c) {
    return __builtin_amdgcn_wmma_f32_16x16x32_bf16(
        false, a, false, b, (short)0, c, false, false);
}

// -------------------------- fp32 -> bf16 convert ---------------------------

__global__ __launch_bounds__(256) void cvt_bf16_kernel(
    const float* __restrict__ src, u16* __restrict__ dst, int n4)
{
    int i = blockIdx.x * 256 + threadIdx.x;
    if (i < n4) {
        float4 v = ((const float4*)src)[i];
        uint2 o;
        o.x = (unsigned)f32_to_bf16(v.x) | ((unsigned)f32_to_bf16(v.y) << 16);
        o.y = (unsigned)f32_to_bf16(v.z) | ((unsigned)f32_to_bf16(v.w) << 16);
        ((uint2*)dst)[i] = o;
    }
}

// -------------------------- QKV GEMM + RoPE --------------------------------
// grid = (BS/128, NH, 3).  Block computes 128 rows x 128 cols (one head) of
// one of Q/K/V.  Weight tile staged in LDS as [n][k] for contiguous B-frags.
// RoPE applied in-register for Q/K.  Output head-major [B][H][S][HD] bf16.

__global__ __launch_bounds__(256) void qkv_rope_kernel(
    const u16* __restrict__ xb,
    const u16* __restrict__ wq, const u16* __restrict__ wk, const u16* __restrict__ wv,
    u16* __restrict__ Qb, u16* __restrict__ Kb, u16* __restrict__ Vb)
{
    __shared__ __align__(16) u16 Bt[128][32];   // [n][k] 8 KB
    const int wave  = threadIdx.x >> 5;
    const int lane  = threadIdx.x & 31;
    const int moff  = (lane >> 4) * 8;
    const int which = blockIdx.z;
    const u16* __restrict__ W  = (which == 0) ? wq : (which == 1) ? wk : wv;
    u16* __restrict__ Out      = (which == 0) ? Qb : (which == 1) ? Kb : Vb;
    const int row0 = blockIdx.x * 128 + wave * 16;
    const int head = blockIdx.y;
    const int n0   = head * HD;

    v8f acc[8] = {};
    for (int k0 = 0; k0 < D_IN; k0 += 32) {
        __syncthreads();
        #pragma unroll
        for (int rep = 0; rep < 2; rep++) {         // stage W[k0..31][n0..127]^T
            int idx = rep * 256 + threadIdx.x;
            int kk = idx >> 4, ng = idx & 15;
            union { uint4 u; u16 s[8]; } t;
            t.u = *(const uint4*)(W + (size_t)(k0 + kk) * D_OUT + n0 + ng * 8);
            #pragma unroll
            for (int e = 0; e < 8; e++) Bt[ng * 8 + e][kk] = t.s[e];
        }
        __syncthreads();
        v16bf a = load_a_frag(xb + (size_t)row0 * D_IN + k0, D_IN, lane);
        #pragma unroll
        for (int j = 0; j < 8; j++) {
            v16bf b = load_b_frag(&Bt[j * 16][0], 32, 0, lane);
            acc[j] = wmma_bf16(a, b, acc[j]);
        }
    }

    if (which < 2) {   // RoPE: pair (d, d+64) lives in acc[j] / acc[j+4]
        #pragma unroll
        for (int j = 0; j < 4; j++) {
            float dcol = (float)(j * 16 + (lane & 15));
            float invf = __expf(dcol * -0.14391157f);   // ln(10000)/64
            #pragma unroll
            for (int r = 0; r < 8; r++) {
                float pos = (float)((row0 + moff + r) & (CTX - 1));
                float s, c;
                __sincosf(pos * invf, &s, &c);
                float x1 = acc[j][r], x2 = acc[j + 4][r];
                acc[j][r]     = x1 * c - x2 * s;
                acc[j + 4][r] = x2 * c + x1 * s;
            }
        }
    }

    #pragma unroll
    for (int j = 0; j < 8; j++) {
        #pragma unroll
        for (int r = 0; r < 8; r++) {
            int rr = row0 + moff + r;
            int bb = rr >> 11, ss = rr & (CTX - 1);
            size_t o = (((size_t)bb * NH + head) * CTX + ss) * HD + j * 16 + (lane & 15);
            Out[o] = f32_to_bf16(acc[j][r]);
        }
    }
}

// -------------------------- causal flash attention -------------------------
// grid = (CTX/128, NH, BATCH).  Block handles 128 query rows of one (b,h).
// Online softmax over 128-key blocks.  Scores NOT scaled (matches reference).

__global__ __launch_bounds__(256) void attn_kernel(
    const u16* __restrict__ Qb, const u16* __restrict__ Kb,
    const u16* __restrict__ Vb, u16* __restrict__ Ctx)
{
    extern __shared__ __align__(16) u16 smem[];
    u16* Kl = smem;                     // [key][d]  128x128 (32 KB)
    u16* Vt = smem + 128 * 128;        // [d][key]  128x128 (32 KB)
    u16* Pl = smem + 2 * 128 * 128;    // [wave][16][32]    (8 KB)

    const int wave = threadIdx.x >> 5;
    const int lane = threadIdx.x & 31;
    const int moff = (lane >> 4) * 8;
    const int qb   = blockIdx.x;
    const int head = blockIdx.y;
    const int b    = blockIdx.z;
    const u16* Qh = Qb + (((size_t)b * NH + head) * CTX) * HD;
    const u16* Kh = Kb + (((size_t)b * NH + head) * CTX) * HD;
    const u16* Vh = Vb + (((size_t)b * NH + head) * CTX) * HD;
    const int q0 = qb * 128 + wave * 16;
    u16* Pw = Pl + wave * 16 * 32;

    v16bf qf[4];
    #pragma unroll
    for (int t = 0; t < 4; t++)
        qf[t] = load_a_frag(Qh + (size_t)q0 * HD + t * 32, HD, lane);

    v8f o[8] = {};
    float mrow[8], lrow[8];
    #pragma unroll
    for (int r = 0; r < 8; r++) { mrow[r] = -3.0e38f; lrow[r] = 0.f; }

    for (int kb = 0; kb <= qb; kb++) {
        __syncthreads();
        #pragma unroll
        for (int rep = 0; rep < 8; rep++) {         // stage K (row-major) + V (transposed)
            int idx = rep * 256 + threadIdx.x;      // 0..2047
            int key = idx >> 4;
            int dg  = (idx & 15) * 8;
            *(uint4*)(Kl + key * HD + dg) =
                *(const uint4*)(Kh + ((size_t)kb * 128 + key) * HD + dg);
            union { uint4 u; u16 s[8]; } t;
            t.u = *(const uint4*)(Vh + ((size_t)kb * 128 + key) * HD + dg);
            #pragma unroll
            for (int e = 0; e < 8; e++) Vt[(dg + e) * 128 + key] = t.s[e];
        }
        if (kb < qb) {                               // prefetch next K block toward L2
            const u16* nk = Kh + ((size_t)(kb + 1) * 128 + (threadIdx.x >> 1)) * HD;
            __builtin_prefetch(nk, 0, 1);
        }
        __syncthreads();

        v8f sacc[8] = {};                            // S = Q K^T  (16 x 128 strip)
        #pragma unroll
        for (int j = 0; j < 8; j++) {
            #pragma unroll
            for (int t = 0; t < 4; t++) {
                v16bf bk = load_b_frag(Kl + (size_t)j * 16 * HD, HD, t * 32, lane);
                sacc[j] = wmma_bf16(qf[t], bk, sacc[j]);
            }
        }

        if (kb == qb) {                              // causal mask on diagonal block
            #pragma unroll
            for (int j = 0; j < 8; j++) {
                int kpos = kb * 128 + j * 16 + (lane & 15);
                #pragma unroll
                for (int r = 0; r < 8; r++)
                    if (kpos > q0 + moff + r) sacc[j][r] = -3.0e38f;
            }
        }

        #pragma unroll
        for (int r = 0; r < 8; r++) {                // online softmax row state
            float mx = mrow[r];
            #pragma unroll
            for (int j = 0; j < 8; j++) mx = fmaxf(mx, sacc[j][r]);
            #pragma unroll
            for (int m = 1; m < 16; m <<= 1) mx = fmaxf(mx, __shfl_xor(mx, m, 32));
            float alpha = __expf(mrow[r] - mx);
            mrow[r] = mx;
            lrow[r] *= alpha;
            #pragma unroll
            for (int j = 0; j < 8; j++) o[j][r] *= alpha;
            float rs = 0.f;
            #pragma unroll
            for (int j = 0; j < 8; j++) {
                float p = __expf(sacc[j][r] - mx);
                sacc[j][r] = p;
                rs += p;
            }
            #pragma unroll
            for (int m = 1; m < 16; m <<= 1) rs += __shfl_xor(rs, m, 32);
            lrow[r] += rs;
        }

        #pragma unroll
        for (int t = 0; t < 4; t++) {                // O += P V, 32 keys per step
            #pragma unroll
            for (int jj = 0; jj < 2; jj++) {         // C-layout -> A-layout via LDS
                int j = t * 2 + jj;
                #pragma unroll
                for (int r = 0; r < 8; r++)
                    Pw[(moff + r) * 32 + jj * 16 + (lane & 15)] = f32_to_bf16(sacc[j][r]);
            }
            v16bf pa = load_a_frag(Pw, 32, lane);
            #pragma unroll
            for (int j = 0; j < 8; j++) {
                v16bf bv = load_b_frag(Vt + (size_t)j * 16 * 128, 128, t * 32, lane);
                o[j] = wmma_bf16(pa, bv, o[j]);
            }
        }
    }

    #pragma unroll
    for (int j = 0; j < 8; j++) {
        #pragma unroll
        for (int r = 0; r < 8; r++) {
            int qpos = q0 + moff + r;
            float val = o[j][r] / lrow[r];
            Ctx[((size_t)b * CTX + qpos) * D_OUT + head * HD + j * 16 + (lane & 15)]
                = f32_to_bf16(val);
        }
    }
}

// -------------------------- output projection ------------------------------
// grid = (BS/128, D_OUT/128).  ctx(bf16) @ W_out(bf16) -> out(f32)

__global__ __launch_bounds__(256) void proj_kernel(
    const u16* __restrict__ Cb, const u16* __restrict__ wo, float* __restrict__ out)
{
    __shared__ __align__(16) u16 Bt[128][32];
    const int wave = threadIdx.x >> 5;
    const int lane = threadIdx.x & 31;
    const int moff = (lane >> 4) * 8;
    const int row0 = blockIdx.x * 128 + wave * 16;
    const int n0   = blockIdx.y * 128;

    v8f acc[8] = {};
    for (int k0 = 0; k0 < D_OUT; k0 += 32) {
        __syncthreads();
        #pragma unroll
        for (int rep = 0; rep < 2; rep++) {
            int idx = rep * 256 + threadIdx.x;
            int kk = idx >> 4, ng = idx & 15;
            union { uint4 u; u16 s[8]; } t;
            t.u = *(const uint4*)(wo + (size_t)(k0 + kk) * D_OUT + n0 + ng * 8);
            #pragma unroll
            for (int e = 0; e < 8; e++) Bt[ng * 8 + e][kk] = t.s[e];
        }
        __syncthreads();
        v16bf a = load_a_frag(Cb + (size_t)row0 * D_OUT + k0, D_OUT, lane);
        #pragma unroll
        for (int j = 0; j < 8; j++) {
            v16bf b = load_b_frag(&Bt[j * 16][0], 32, 0, lane);
            acc[j] = wmma_bf16(a, b, acc[j]);
        }
    }
    #pragma unroll
    for (int j = 0; j < 8; j++) {
        #pragma unroll
        for (int r = 0; r < 8; r++) {
            int rr = row0 + moff + r;
            out[(size_t)rr * D_OUT + n0 + j * 16 + (lane & 15)] = acc[j][r];
        }
    }
}

// -------------------------- host launcher ----------------------------------

extern "C" void kernel_launch(void* const* d_in, const int* in_sizes, int n_in,
                              void* d_out, int out_size, void* d_ws, size_t ws_size,
                              hipStream_t stream) {
    const float* x  = (const float*)d_in[0];
    const float* Wq = (const float*)d_in[1];
    const float* Wk = (const float*)d_in[2];
    const float* Wv = (const float*)d_in[3];
    const float* Wo = (const float*)d_in[4];
    float* out = (float*)d_out;

    char* ws = (char*)d_ws;
    size_t off = 0;
    auto take = [&](size_t bytes) -> char* {
        char* p = ws + off;
        off += (bytes + 255) & ~(size_t)255;
        return p;
    };
    u16* xb  = (u16*)take((size_t)BS * D_IN * 2);
    u16* wqb = (u16*)take((size_t)D_IN * D_OUT * 2);
    u16* wkb = (u16*)take((size_t)D_IN * D_OUT * 2);
    u16* wvb = (u16*)take((size_t)D_IN * D_OUT * 2);
    u16* wob = (u16*)take((size_t)D_OUT * D_OUT * 2);
    u16* Qb  = (u16*)take((size_t)BATCH * NH * CTX * HD * 2);
    u16* Kb  = (u16*)take((size_t)BATCH * NH * CTX * HD * 2);
    u16* Vb  = (u16*)take((size_t)BATCH * NH * CTX * HD * 2);
    u16* Cb  = (u16*)take((size_t)BS * D_OUT * 2);

    const int nx = BS * D_IN;          // 8388608
    const int nw = D_IN * D_OUT;       // 4194304
    cvt_bf16_kernel<<<nx / 1024, 256, 0, stream>>>(x,  xb,  nx / 4);
    cvt_bf16_kernel<<<nw / 1024, 256, 0, stream>>>(Wq, wqb, nw / 4);
    cvt_bf16_kernel<<<nw / 1024, 256, 0, stream>>>(Wk, wkb, nw / 4);
    cvt_bf16_kernel<<<nw / 1024, 256, 0, stream>>>(Wv, wvb, nw / 4);
    cvt_bf16_kernel<<<nw / 1024, 256, 0, stream>>>(Wo, wob, nw / 4);

    qkv_rope_kernel<<<dim3(BS / 128, NH, 3), 256, 0, stream>>>(
        xb, wqb, wkb, wvb, Qb, Kb, Vb);

    size_t attn_smem = (size_t)(2 * 128 * 128 + 8 * 16 * 32) * sizeof(u16); // 73728 B
    attn_kernel<<<dim3(CTX / 128, NH, BATCH), 256, attn_smem, stream>>>(
        Qb, Kb, Vb, Cb);

    proj_kernel<<<dim3(BS / 128, D_OUT / 128), 256, 0, stream>>>(Cb, wob, out);
}
